// sCIFAR10_GRU_47966194762255
// MI455X (gfx1250) — matrix operand order repaired and checked by reading, not compiled, not run on hardware
//
#include <hip/hip_runtime.h>

#define H        181
#define HP       192
#define T_STEPS  1024
#define BATCH    1024
#define NCLS     10
#define KC_CHUNKS 6
#define NWAVES   12
#define NTHREADS (NWAVES * 32)

typedef __attribute__((ext_vector_type(16))) __bf16 v16bf;
typedef __attribute__((ext_vector_type(8)))  float  v8f;

union Frag16 { v16bf v; uint4 u[2]; };

__device__ __forceinline__ unsigned f2bfbits(float f) {
  union { float f; unsigned u; } x; x.f = f;
  return (x.u + 0x7fffu + ((x.u >> 16) & 1u)) >> 16;   // RNE truncate to bf16
}
__device__ __forceinline__ float bf2f(unsigned short h) {
  union { unsigned u; float f; } x; x.u = ((unsigned)h) << 16;
  return x.f;
}
__device__ __forceinline__ float fast_sigmoid(float v) {
  float e = __expf(-v);
  return __builtin_amdgcn_rcpf(1.0f + e);
}
__device__ __forceinline__ float fast_tanh(float v) {
  float e = __expf(-2.0f * v);
  return (1.0f - e) * __builtin_amdgcn_rcpf(1.0f + e);
}

extern "C" __global__ void __launch_bounds__(NTHREADS, 1)
gru_persistent_kernel(const float* __restrict__ x,
                      const float* __restrict__ w_ih,
                      const float* __restrict__ w_hh,
                      const float* __restrict__ b_ih,
                      const float* __restrict__ b_hh,
                      const float* __restrict__ w_fc,
                      const float* __restrict__ b_fc,
                      float* __restrict__ out) {
  // Only the hidden-state ping/pong buffer lives in LDS (12 KB).
  __shared__ unsigned short hb[2][16 * HP];

  const int tid  = threadIdx.x;
  const int lane = tid & 31;
  const int wave = tid >> 5;          // = column tile index jc (0..11)
  const int rowbase = blockIdx.x * 16;
  const int lo16 = lane & 15;
  const bool hi  = lane >= 16;

  // ---- load this wave's slice of w_hh ONCE into persistent registers, in
  // a contiguous-K B fragment order (same K order as A). Fragment (g = gate, kc = K-chunk): lane = column
  // (lo16); lanes>=16 take K base +16; dword v = packed K pair (2v, 2v+1).
  // 18 fragments * 8 VGPRs = 144 VGPRs -> whole wave stays under 256 VGPRs,
  // so no s_set_vgpr_msb window switching and 3 waves/SIMD occupancy.
  Frag16 bfr[3][KC_CHUNKS];
  {
    const int ncol = wave * 16 + lo16;
    const bool nval = ncol < H;
    const int kb_ = hi ? 16 : 0;
    #pragma unroll
    for (int g = 0; g < 3; ++g) {
      const float* wrow = w_hh + ((size_t)g * H + ncol) * H;  // B[k][n] = w_hh[n][k]
      #pragma unroll
      for (int kc = 0; kc < KC_CHUNKS; ++kc) {
        unsigned dw[8];
        #pragma unroll
        for (int e = 0; e < 8; ++e) {
          int k0 = kc * 32 + kb_ + 2 * e;
          float f0 = (nval && (k0     < H)) ? wrow[k0]     : 0.0f;
          float f1 = (nval && (k0 + 1 < H)) ? wrow[k0 + 1] : 0.0f;
          dw[e] = f2bfbits(f0) | (f2bfbits(f1) << 16);
        }
        uint4 q0; q0.x = dw[0]; q0.y = dw[1]; q0.z = dw[2]; q0.w = dw[3];
        uint4 q1; q1.x = dw[4]; q1.y = dw[5]; q1.z = dw[6]; q1.w = dw[7];
        bfr[g][kc].u[0] = q0;
        bfr[g][kc].u[1] = q1;
      }
    }
  }

  // zero both h buffers (h0 = 0)
  {
    unsigned* hz = (unsigned*)hb;
    for (int i = tid; i < (int)(sizeof(hb) / 4); i += NTHREADS) hz[i] = 0u;
  }

  // ---- per-column gate constants (w_ih / biases) ----
  float wir, wiz, win, br_, bz_, bni, bnh;
  {
    int c  = wave * 16 + lo16;
    bool cv = c < H;
    wir = cv ? w_ih[c]         : 0.f;
    wiz = cv ? w_ih[H + c]     : 0.f;
    win = cv ? w_ih[2 * H + c] : 0.f;
    br_ = cv ? (b_ih[c]     + b_hh[c])     : 0.f;
    bz_ = cv ? (b_ih[H + c] + b_hh[H + c]) : 0.f;
    bni = cv ? b_ih[2 * H + c] : 0.f;
    bnh = cv ? b_hh[2 * H + c] : 0.f;
  }

  float hreg[8];                     // persistent fp32 hidden state (C layout)
  #pragma unroll
  for (int v = 0; v < 8; ++v) hreg[v] = 0.f;

  const int arow   = lo16;           // A: lane holds row lane%16
  const int akoff  = hi ? 16 : 0;    // A: upper lanes start at K+16
  const int rowoff = hi ? 8 : 0;     // C/D: dword v = row v (lo) / v+8 (hi)
  const int col    = wave * 16 + lo16;

  __syncthreads();

  // One GRU step: read h from hr, write new h to hw. B comes from registers.
  auto gru_step = [&](const unsigned short* __restrict__ hr,
                      unsigned short* __restrict__ hw,
                      const float* __restrict__ xv) {
    auto loadA = [&](Frag16& f, int kc) {
      const uint4* p = (const uint4*)((const char*)hr +
                        (size_t)(arow * HP + kc * 32 + akoff) * 2);
      f.u[0] = p[0];
      f.u[1] = p[1];   // +8 elements = +16 bytes
    };
    Frag16 afb[2];                   // 2-deep A prefetch: keeps VGPRs < 256
    loadA(afb[0], 0);
    v8f ar = {0.f,0.f,0.f,0.f,0.f,0.f,0.f,0.f};
    v8f az = {0.f,0.f,0.f,0.f,0.f,0.f,0.f,0.f};
    v8f an = {0.f,0.f,0.f,0.f,0.f,0.f,0.f,0.f};
    #pragma unroll
    for (int kc = 0; kc < KC_CHUNKS; ++kc) {
      if (kc + 1 < KC_CHUNKS) loadA(afb[(kc + 1) & 1], kc + 1);
      const v16bf a = afb[kc & 1].v;
      ar = __builtin_amdgcn_wmma_f32_16x16x32_bf16(false, a, false, bfr[0][kc].v, (short)0, ar, false, false);
      az = __builtin_amdgcn_wmma_f32_16x16x32_bf16(false, a, false, bfr[1][kc].v, (short)0, az, false, false);
      an = __builtin_amdgcn_wmma_f32_16x16x32_bf16(false, a, false, bfr[2][kc].v, (short)0, an, false, false);
    }
    // fused gate math entirely in accumulator registers
    #pragma unroll
    for (int v = 0; v < 8; ++v) {
      float pr_ = ar[v] + xv[v] * wir + br_;
      float pz_ = az[v] + xv[v] * wiz + bz_;
      float rg  = fast_sigmoid(pr_);
      float zg  = fast_sigmoid(pz_);
      float pn_ = xv[v] * win + bni + rg * (an[v] + bnh);
      float ng  = fast_tanh(pn_);
      float hn  = (1.f - zg) * ng + zg * hreg[v];
      hreg[v] = hn;
      hw[(v + rowoff) * HP + col] = (unsigned short)f2bfbits(hn);
    }
  };

  // ============ sequential scan, unrolled x2 (explicit ping/pong) ==========
  for (int t = 0; t < T_STEPS; t += 2) {
    // fetch x for both sub-steps with one b64 per row (uniform per half-wave)
    float xe[8], xo[8];
    #pragma unroll
    for (int v = 0; v < 8; ++v) {
      const float2 xl = *(const float2*)(x + (size_t)(rowbase + v + rowoff) * T_STEPS + t);
      xe[v] = xl.x;
      xo[v] = xl.y;
    }
    gru_step(hb[0], hb[1], xe);
    __syncthreads();
    gru_step(hb[1], hb[0], xo);
    __syncthreads();
  }

  // ---- final FC: logits = h_last @ w_fc^T + b_fc (h_last is buffer 0) ----
  const unsigned short* hf = hb[0];
  for (int p = tid; p < 16 * NCLS; p += NTHREADS) {
    int row = p / NCLS, cls = p % NCLS;
    float acc = b_fc[cls];
    for (int k = 0; k < H; ++k)
      acc += bf2f(hf[row * HP + k]) * w_fc[cls * H + k];
    out[(size_t)(rowbase + row) * NCLS + cls] = acc;
  }
}

extern "C" void kernel_launch(void* const* d_in, const int* in_sizes, int n_in,
                              void* d_out, int out_size, void* d_ws, size_t ws_size,
                              hipStream_t stream) {
  (void)in_sizes; (void)n_in; (void)out_size; (void)d_ws; (void)ws_size;
  const float* x    = (const float*)d_in[0];
  const float* w_ih = (const float*)d_in[1];
  const float* w_hh = (const float*)d_in[2];
  const float* b_ih = (const float*)d_in[3];
  const float* b_hh = (const float*)d_in[4];
  const float* w_fc = (const float*)d_in[5];
  const float* b_fc = (const float*)d_in[6];
  float* out = (float*)d_out;

  gru_persistent_kernel<<<BATCH / 16, NTHREADS, 0, stream>>>(
      x, w_ih, w_hh, b_ih, b_hh, w_fc, b_fc, out);
}
